// EmbeddingBlock_64080912056984
// MI455X (gfx1250) — matrix operand order repaired and checked by reading, not compiled, not run on hardware
//
#include <hip/hip_runtime.h>
#include <cstdint>
#include <cstddef>

// ---------------- problem constants (fixed by the reference) ----------------
#define BATCH   8
#define NNODE   2048
#define NEDGE   65536
#define HDIM    128
#define NRAD    6
#define TILE_M  128
#define EPB     (NEDGE / TILE_M)     // 512 edge-tiles per batch
#define NBLOCKS (BATCH * EPB)        // 4096 workgroups

// ---------------- WMMA / TDM vector types (gfx1250, wave32) ----------------
typedef __attribute__((ext_vector_type(16))) __bf16 v16bf;
typedef __attribute__((ext_vector_type(8)))  float  v8f;
typedef __attribute__((ext_vector_type(4)))  unsigned int u32x4;
typedef __attribute__((ext_vector_type(8)))  int    i32x8;
typedef __attribute__((ext_vector_type(4)))  int    i32x4;

#if defined(__gfx1250__) && __has_builtin(__builtin_amdgcn_tensor_load_to_lds) && \
    __has_builtin(__builtin_amdgcn_s_wait_tensorcnt)
#define USE_TDM 1
#else
#define USE_TDM 0
#endif

union Frag16 {
    unsigned short u[16];
    v16bf          v;
};

__device__ __forceinline__ unsigned short f2bf(float f) {
    unsigned int u = __float_as_uint(f);
    unsigned int r = 0x7FFFu + ((u >> 16) & 1u);   // round-to-nearest-even
    return (unsigned short)((u + r) >> 16);
}

// silu via fast v_rcp_f32 (avoid the IEEE division macro-expansion:
// v_div_scale/v_div_fmas chains cost ~10 VALU each; we do 128 silu per lane)
__device__ __forceinline__ float silu(float x) {
    return x * __builtin_amdgcn_rcpf(1.0f + __expf(-x));
}

// ---------------------------------------------------------------------------
// Prep kernel: fold the rank-1 embedding through lin_w, convert the r-block
// of lin_w to bf16.  Workspace layout:
//   [0, 32768)         : W3 bf16, row-major [o=128][k=128]  (lin_w[:,256:384])
//   [32768, 33280)     : w1[128]  = lin_w[:,0:128]   @ emb_w
//   [33280, 33792)     : w2[128]  = lin_w[:,128:256] @ emb_w
//   [33792, 34304)     : c[128]   = lin_b + lin_w[:,0:256] @ [emb_b;emb_b]
// ---------------------------------------------------------------------------
__global__ __launch_bounds__(128) void prep_kernel(
    const float* __restrict__ emb_w, const float* __restrict__ emb_b,
    const float* __restrict__ lin_w, const float* __restrict__ lin_b,
    unsigned short* __restrict__ W3,
    float* __restrict__ w1, float* __restrict__ w2, float* __restrict__ cvec)
{
    const int o = threadIdx.x;           // 0..127
    const float* lw = lin_w + (size_t)o * (3 * HDIM);
    float s1 = 0.f, s2 = 0.f, sc = 0.f;
    #pragma unroll 4
    for (int k = 0; k < HDIM; ++k) {
        float a = lw[k];
        float bgl = lw[HDIM + k];
        float ew = emb_w[k];
        s1 += a * ew;
        s2 += bgl * ew;
        sc += (a + bgl) * emb_b[k];
    }
    w1[o] = s1;
    w2[o] = s2;
    cvec[o] = lin_b[o] + sc;
    unsigned short* w3r = W3 + o * HDIM;
    #pragma unroll 4
    for (int k = 0; k < HDIM; ++k)
        w3r[k] = f2bf(lw[2 * HDIM + k]);
}

// ---------------------------------------------------------------------------
// Fused edge kernel: one workgroup = 128 edges x 128 outputs of one batch.
//   r computed directly into WMMA A-fragment layout (registers only).
//   B tile (bf16 W3) staged once into padded LDS via the Tensor Data Mover
//   (TDM pad_interval/pad_amount reproduce the +8-half row padding exactly).
//   Epilogue adds folded rank-1 terms, silu, streams rows with NT stores.
// ---------------------------------------------------------------------------
__global__ __launch_bounds__(256) void fused_edge_kernel(
    const float* __restrict__ x,      // [B, N]
    const float* __restrict__ rbf,    // [B, E, 6]
    const int*   __restrict__ ei,     // [E]
    const int*   __restrict__ ej,     // [E]
    const float* __restrict__ rbf_w,  // [128, 6]
    const float* __restrict__ rbf_b,  // [128]
    const unsigned short* __restrict__ W3,   // bf16 [128][128]
    const float* __restrict__ w1,
    const float* __restrict__ w2,
    const float* __restrict__ cvec,
    float* __restrict__ out)          // [B, E, 128]
{
    __shared__ unsigned short sB[HDIM][136];  // padded: +8 halves -> 4-bank row shift
    __shared__ float s_rbfw[HDIM][8];
    __shared__ float s_rbfb[HDIM];
    __shared__ float s_w1[HDIM], s_w2[HDIM], s_c[HDIM];
    __shared__ float s_xi[TILE_M], s_xj[TILE_M];

    const int t    = threadIdx.x;
    const int lane = t & 31;
    const int wv   = t >> 5;          // 8 waves
    const int g    = lane >> 4;       // half-wave group
    const int l16  = lane & 15;

    const int b  = blockIdx.x / EPB;
    const int e0 = (blockIdx.x % EPB) * TILE_M;

    // ---- stage B tile (bf16 weights) into padded LDS ----
#if USE_TDM
    if (wv == 0) {
        // Tensor DMA descriptor: 2D tensor 128x128 x 2B, contiguous rows in
        // global, +4 DWORDs of LDS padding after every 64 DWORDs (one row).
        const unsigned lds_off =
            (unsigned)(size_t)(__attribute__((address_space(3))) unsigned short*)&sB[0][0];
        const unsigned long long ga = (unsigned long long)(size_t)W3;

        u32x4 g0 = {};
        g0[0] = 1u;                                   // count=1, user load descriptor
        g0[1] = lds_off;                              // lds_addr
        g0[2] = (unsigned)(ga & 0xFFFFFFFFu);         // global_addr[31:0]
        g0[3] = (unsigned)((ga >> 32) & 0x01FFFFFFu)  // global_addr[56:32]
              | (2u << 30);                           // type=2 (image)

        i32x8 g1 = {};
        g1[0] = (1 << 16)      // data_size = 2 bytes
              | (1 << 20)      // pad_enable
              | (5 << 22)      // pad_interval: 64 DWORDs (one 128-half row)
              | (3 << 25);     // pad_amount: 4 DWORDs (8 halves)
        g1[1] = (HDIM << 16);  // tensor_dim0 = 128  (bits 79:48, low half)
        g1[2] = (HDIM << 16);  // tensor_dim0 hi=0 | tensor_dim1 = 128 (low half)
        g1[3] = (HDIM << 16);  // tensor_dim1 hi=0 | tile_dim0 = 128
        g1[4] = HDIM;          // tile_dim1 = 128 | tile_dim2 = 0
        g1[5] = HDIM;          // tensor_dim0_stride = 128 elements
        g1[6] = 0;             // tensor_dim1_stride = 0 (2D)
        g1[7] = 0;

        i32x4 z4 = {};
#if __clang_major__ >= 23
        i32x8 z8 = {};
        __builtin_amdgcn_tensor_load_to_lds(g0, g1, z4, z4, z8, 0);
#else
        __builtin_amdgcn_tensor_load_to_lds(g0, g1, z4, z4, 0);
#endif
        __builtin_amdgcn_s_wait_tensorcnt(0);
    }
#else
    {
        const int n  = t >> 1;
        const int k0 = (t & 1) * 64;
        const uint4* src = (const uint4*)(W3 + n * HDIM + k0);
        #pragma unroll
        for (int q = 0; q < 8; ++q)
            *(uint4*)&sB[n][k0 + q * 8] = src[q];
    }
#endif

    // ---- stage small constants & gather x[i], x[j] ----
    if (t < HDIM) {
        s_rbfb[t] = rbf_b[t];
        s_w1[t] = w1[t];
        s_w2[t] = w2[t];
        s_c[t]  = cvec[t];
        #pragma unroll
        for (int q = 0; q < NRAD; ++q)
            s_rbfw[t][q] = rbf_w[t * NRAD + q];
        s_xi[t] = x[b * NNODE + ei[e0 + t]];
    } else {
        const int tt = t - HDIM;
        s_xj[tt] = x[b * NNODE + ej[e0 + tt]];
    }

    // ---- per-lane edge-row rbf values (row matches A-fragment M index) ----
    const int mrow = (wv << 4) + l16;                // row in tile this lane feeds
    const float* rp = rbf + (size_t)(b * NEDGE + e0 + mrow) * NRAD;
    __builtin_prefetch(rp, 0, 0);                    // global_prefetch_b8
    float rb[NRAD];
    #pragma unroll
    for (int q = 0; q < NRAD; ++q) rb[q] = rp[q];

    __syncthreads();

    v8f acc[8] = {};   // 8 column tiles of 16x16 f32

    #pragma unroll
    for (int kb = 0; kb < 4; ++kb) {
        // Build bf16 A fragment in-register: lane holds rows M=l16 of 16x32 tile,
        // K(p,h,g) = 16*(p>>2) + 8*g + 2*(p&3) + h   (ISA 16-bit A layout)
        Frag16 af;
        #pragma unroll
        for (int p = 0; p < 8; ++p) {
            #pragma unroll
            for (int h = 0; h < 2; ++h) {
                const int k = (kb << 5) + ((p >> 2) << 4) + (g << 3) + ((p & 3) << 1) + h;
                float a = s_rbfb[k];
                #pragma unroll
                for (int q = 0; q < NRAD; ++q)
                    a += rb[q] * s_rbfw[k][q];
                af.u[p * 2 + h] = f2bf(silu(a));
            }
        }
        // 8 column tiles: B fragment = halves [8g,8g+8) and [16+8g,24+8g) of row n
        #pragma unroll
        for (int nb = 0; nb < 8; ++nb) {
            const int n = (nb << 4) + l16;
            Frag16 bf;
            *(uint4*)&bf.u[0] = *(const uint4*)&sB[n][(kb << 5) + (g << 3)];
            *(uint4*)&bf.u[8] = *(const uint4*)&sB[n][(kb << 5) + (g << 3) + 16];
            acc[nb] = __builtin_amdgcn_wmma_f32_16x16x32_bf16(
                false, af.v, false, bf.v, (short)0, acc[nb], false, false);
        }
    }

    // ---- epilogue: rank-1 folded terms + silu, stream rows (NT stores) ----
    #pragma unroll
    for (int v = 0; v < 8; ++v) {
        const int row = (wv << 4) + v + (g << 3);    // C/D layout: M = v (+8 for g=1)
        const float xiv = s_xi[row];
        const float xjv = s_xj[row];
        float* orow = out + (size_t)(b * NEDGE + e0 + row) * HDIM;
        #pragma unroll
        for (int nb = 0; nb < 8; ++nb) {
            const int col = (nb << 4) + l16;
            float o = acc[nb][v] + xiv * s_w1[col] + xjv * s_w2[col] + s_c[col];
            __builtin_nontemporal_store(silu(o), orow + col);
        }
    }
}

// ---------------------------------------------------------------------------
extern "C" void kernel_launch(void* const* d_in, const int* in_sizes, int n_in,
                              void* d_out, int out_size, void* d_ws, size_t ws_size,
                              hipStream_t stream) {
    const float* x     = (const float*)d_in[0];
    const float* rbf   = (const float*)d_in[1];
    const int*   ei    = (const int*)d_in[2];
    const int*   ej    = (const int*)d_in[3];
    const float* emb_w = (const float*)d_in[4];
    const float* emb_b = (const float*)d_in[5];
    const float* rbf_w = (const float*)d_in[6];
    const float* rbf_b = (const float*)d_in[7];
    const float* lin_w = (const float*)d_in[8];
    const float* lin_b = (const float*)d_in[9];

    unsigned short* W3 = (unsigned short*)d_ws;                    // 32768 B
    float* w1 = (float*)((char*)d_ws + 32768);                     // 512 B
    float* w2 = w1 + HDIM;                                         // 512 B
    float* cv = w2 + HDIM;                                         // 512 B

    prep_kernel<<<1, 128, 0, stream>>>(emb_w, emb_b, lin_w, lin_b, W3, w1, w2, cv);

    fused_edge_kernel<<<NBLOCKS, 256, 0, stream>>>(
        x, rbf, ei, ej, rbf_w, rbf_b, W3, w1, w2, cv, (float*)d_out);
}
